// AtomAttentionEncoder_24412594111347
// MI455X (gfx1250) — matrix operand order repaired and checked by reading, not compiled, not run on hardware
//
#include <hip/hip_runtime.h>
#include <hip/hip_bf16.h>
#include <math.h>

// ---------------------------------------------------------------------------
// AtomAttentionEncoder for MI455X (gfx1250, wave32, WMMA 16x16x32 f16->f32)
// Round 4: fix async builtin signature (takes generic int4* pointers).
// ---------------------------------------------------------------------------

typedef __attribute__((ext_vector_type(16))) _Float16 v16h;
typedef __attribute__((ext_vector_type(8)))  float    v8f;
typedef __attribute__((ext_vector_type(4)))  int      v4i_t;

struct __align__(8) h4 { _Float16 x, y, z, w; };

#define N_ATOM  8192
#define N_TOKEN 2048
#define NQ      32
#define NK      128
#define NBLOCKS 256      // N_ATOM / NQ
#define NHEAD   4
#define CH      32
#define NBLK    3
#define KCAT    416      // 3 + 1 + 1 + 128 + 256 + pad -> 13 * 32

// ---- CDNA5 async global->LDS (ASYNCcnt) probe-guarded ----------------------
#ifndef __has_builtin
#define __has_builtin(x) 0
#endif
#if __has_builtin(__builtin_amdgcn_global_load_async_to_lds_b128) && \
    __has_builtin(__builtin_amdgcn_s_wait_asynccnt)
#define HAVE_ASYNC_LDS 1
#else
#define HAVE_ASYNC_LDS 0
#endif

#if HAVE_ASYNC_LDS
__device__ __forceinline__ void async_b128(const void* g, void* l) {
  __builtin_amdgcn_global_load_async_to_lds_b128((v4i_t*)g, (v4i_t*)l, 0, 0);
}
#endif

// ------------------------- WMMA fragment helpers ---------------------------
// A fragment (16x32 f16): lane L<16 -> row M=L, K in {0..7, 16..23};
//                         lane L>=16 -> row M=L-16, K in {8..15, 24..31}.
// B fragment (32x16 f16): lane L -> col N=L&15, K same split as A fragment.
// C/D (16x16 f32): lane L -> col N=L&15, rows (L<16 ? 0..7 : 8..15).

__device__ __forceinline__ v8f wmma_f16(v16h a, v16h b, v8f c) {
  return __builtin_amdgcn_wmma_f32_16x16x32_f16(false, a, false, b, (short)0, c,
                                                false, false);
}

// A from f16 LDS, row-major [rows][ldA], K runs contiguous -> 2x ds_load_b128.
__device__ __forceinline__ v16h frag_a_h(const _Float16* A, int ldA,
                                         int m0, int k0, int lane) {
  int row = m0 + (lane & 15);
  int kb = (lane < 16) ? 0 : 8;
  v16h a;
#pragma unroll
  for (int i = 0; i < 16; ++i) {
    int k = k0 + kb + i + ((i >= 8) ? 8 : 0);
    a[i] = A[row * ldA + k];
  }
  return a;
}

// B from f16 LDS stored TRANSPOSED as Bt[col][k] -> two 16B runs (same K split as A).
__device__ __forceinline__ v16h frag_b_t(const _Float16* Bt, int ldK,
                                         int k0, int n0, int lane) {
  int col = n0 + (lane & 15);
  int ks = k0 + ((lane < 16) ? 0 : 8);
  v16h b;
#pragma unroll
  for (int i = 0; i < 16; ++i) b[i] = Bt[col * ldK + ks + i + ((i >= 8) ? 8 : 0)];
  return b;
}

// A from f32 buffer (LDS softmax tile): contiguous float4 runs + cvt.
__device__ __forceinline__ v16h frag_a_f32(const float* A, int ldA,
                                           int m0, int k0, int lane) {
  int row = m0 + (lane & 15);
  int kb = (lane < 16) ? 0 : 8;
  v16h a;
#pragma unroll
  for (int i = 0; i < 16; ++i) {
    int k = k0 + kb + i + ((i >= 8) ? 8 : 0);
    a[i] = (_Float16)A[(size_t)row * ldA + k];
  }
  return a;
}

template <int ACT>
__device__ __forceinline__ float act_apply(float v) {
  if (ACT == 1) v = fmaxf(v, 0.f);
  if (ACT == 2) v = 1.f / (1.f + __expf(-v));
  if (ACT == 3) v = v / (1.f + __expf(-v));
  return v;
}

// ------------------------- generic WMMA GEMM -------------------------------
// act(A[M,K] @ W[K,N]) (+resid) -> out (f32), out_b (f32 copy), out16 (f16).
// Requirements: M%128==0, N%32==0, K%32==0. Workgroup = 8 waves = 128x32
// macro-tile; per K-step stage A(128x32) + W(32x32 transposed) in LDS f16.
template <int ACT>
__global__ __launch_bounds__(256) void gemm_wmma(
    const float* __restrict__ A, const float* __restrict__ W,
    const float* resid, float* out, float* out_b, _Float16* out16,
    int M, int N, int K) {
  __shared__ __align__(16) _Float16 ldsA[128 * 32];
  __shared__ __align__(16) _Float16 ldsW[32 * 32];    // [col][k]
  __shared__ __align__(16) float    ldsOut[8][256];

  const int tid = threadIdx.x;
  const int wave = tid >> 5, lane = tid & 31;
  const int nWgN = N >> 5;
  const int mWg = blockIdx.x / nWgN, nWg = blockIdx.x % nWgN;
  const int m0wg = mWg << 7, n0wg = nWg << 5;
  const int m0 = m0wg + (wave << 4);

  const int arow = tid >> 3;         // 0..31  (A stage: +32 per pass)
  const int akg = (tid & 7) << 2;    // 0,4,...,28

  v8f acc0 = {0.f, 0.f, 0.f, 0.f, 0.f, 0.f, 0.f, 0.f};
  v8f acc1 = {0.f, 0.f, 0.f, 0.f, 0.f, 0.f, 0.f, 0.f};

  for (int k0 = 0; k0 < K; k0 += 32) {
    // stage A tile (coalesced float4 loads, packed half stores)
#pragma unroll
    for (int p = 0; p < 4; ++p) {
      int row = (p << 5) + arow;
      float4 va = *(const float4*)(A + (size_t)(m0wg + row) * K + k0 + akg);
      h4 hv = {(_Float16)va.x, (_Float16)va.y, (_Float16)va.z, (_Float16)va.w};
      *(h4*)(ldsA + row * 32 + akg) = hv;
    }
    // stage W tile transposed: ldsW[col][k]
#pragma unroll
    for (int p = 0; p < 4; ++p) {
      int idx = (p << 8) + tid;
      int kk = idx >> 5, cc = idx & 31;
      ldsW[cc * 32 + kk] = (_Float16)W[(size_t)(k0 + kk) * N + n0wg + cc];
    }
    __syncthreads();
    if (k0 + 32 < K)
      __builtin_prefetch(A + (size_t)(m0wg + arow) * K + k0 + 32 + akg, 0, 1);
    v16h af = frag_a_h(ldsA, 32, wave << 4, 0, lane);
    v16h b0 = frag_b_t(ldsW, 32, 0, 0, lane);
    v16h b1 = frag_b_t(ldsW, 32, 0, 16, lane);
    acc0 = wmma_f16(af, b0, acc0);
    acc1 = wmma_f16(af, b1, acc1);
    __syncthreads();
  }

  // epilogue: stage 16x16 f32 tile per wave in LDS, store coalesced float4s
  const int col = lane & 15;
  const int rb = (lane < 16) ? 0 : 8;
  v8f accs[2] = {acc0, acc1};
#pragma unroll
  for (int half = 0; half < 2; ++half) {
    v8f acc = accs[half];
    int n0 = n0wg + (half << 4);
#pragma unroll
    for (int r = 0; r < 8; ++r) ldsOut[wave][(rb + r) * 16 + col] = acc[r];
    // per-wave region, same-wave DS ops are in order -> no block sync needed
#pragma unroll
    for (int qd = lane; qd < 64; qd += 32) {
      int row = qd >> 2, part = (qd & 3) << 2;
      float4 v = *(const float4*)(&ldsOut[wave][row * 16 + part]);
      v.x = act_apply<ACT>(v.x);
      v.y = act_apply<ACT>(v.y);
      v.z = act_apply<ACT>(v.z);
      v.w = act_apply<ACT>(v.w);
      size_t base = (size_t)(m0 + row) * N + n0 + part;
      if (resid) {
        float4 rr = *(const float4*)(resid + base);
        v.x += rr.x; v.y += rr.y; v.z += rr.z; v.w += rr.w;
      }
      if (out) *(float4*)(out + base) = v;
      if (out_b) *(float4*)(out_b + base) = v;
      if (out16) {
        h4 hv = {(_Float16)v.x, (_Float16)v.y, (_Float16)v.z, (_Float16)v.w};
        *(h4*)(out16 + base) = hv;
      }
    }
  }
}

// ------------------------- input concat for embedding ----------------------
__global__ void fill_acat(const float* __restrict__ pos,
                          const float* __restrict__ charge,
                          const float* __restrict__ mask,
                          const float* __restrict__ elem,
                          const float* __restrict__ chars,
                          float* __restrict__ acat) {
  int i = blockIdx.x * blockDim.x + threadIdx.x;
  if (i >= N_ATOM * KCAT) return;
  int row = i / KCAT, col = i % KCAT;
  float v;
  if (col < 3)        v = pos[row * 3 + col];
  else if (col == 3)  v = asinhf(charge[row]);
  else if (col == 4)  v = mask[row];
  else if (col < 133) v = elem[(size_t)row * 128 + (col - 5)];
  else if (col < 389) v = chars[(size_t)row * 256 + (col - 133)];
  else                v = 0.f;
  acat[i] = v;
}

__global__ void fill_wcat(const float* __restrict__ wpos,
                          const float* __restrict__ wch,
                          const float* __restrict__ wmk,
                          const float* __restrict__ wel,
                          const float* __restrict__ wchr,
                          float* __restrict__ wcat) {
  int i = blockIdx.x * blockDim.x + threadIdx.x;
  if (i >= KCAT * 128) return;
  int r = i / 128, c = i % 128;
  float v;
  if (r < 3)        v = wpos[r * 128 + c];
  else if (r == 3)  v = wch[c];
  else if (r == 4)  v = wmk[c];
  else if (r < 133) v = wel[(r - 5) * 128 + c];
  else if (r < 389) v = wchr[(r - 133) * 128 + c];
  else              v = 0.f;
  wcat[i] = v;
}

// pack W_cq | W_ck into [128,32] so the projection is one N=32 GEMM
__global__ void fill_wqk(const float* __restrict__ wcq,
                         const float* __restrict__ wck,
                         float* __restrict__ o) {
  int i = blockIdx.x * blockDim.x + threadIdx.x;
  if (i >= 128 * 32) return;
  int r = i >> 5, c = i & 31;
  o[i] = (c < 16) ? wcq[r * 16 + c] : wck[r * 16 + (c - 16)];
}

// ------------------------- layernorm (wave per row) ------------------------
__global__ __launch_bounds__(256) void ln_kernel(const float* __restrict__ x,
                                                 const float* __restrict__ gamma,
                                                 float* __restrict__ y) {
  const int wave = threadIdx.x >> 5, lane = threadIdx.x & 31;
  const int row = blockIdx.x * 8 + wave;
  const float* xr = x + (size_t)row * 128;
  float v0 = xr[lane], v1 = xr[lane + 32], v2 = xr[lane + 64], v3 = xr[lane + 96];
  float s = v0 + v1 + v2 + v3;
#pragma unroll
  for (int m = 16; m >= 1; m >>= 1) s += __shfl_xor(s, m, 32);
  float mean = s * (1.f / 128.f);
  float d0 = v0 - mean, d1 = v1 - mean, d2 = v2 - mean, d3 = v3 - mean;
  float q = d0 * d0 + d1 * d1 + d2 * d2 + d3 * d3;
#pragma unroll
  for (int m = 16; m >= 1; m >>= 1) q += __shfl_xor(q, m, 32);
  float inv = rsqrtf(q * (1.f / 128.f) + 1e-5f);
  float* yr = y + (size_t)row * 128;
  yr[lane]      = d0 * inv * gamma[lane];
  yr[lane + 32] = d1 * inv * gamma[lane + 32];
  yr[lane + 64] = d2 * inv * gamma[lane + 64];
  yr[lane + 96] = d3 * inv * gamma[lane + 96];
}

// ------------------------- fused pair tensor kernel ------------------------
// One workgroup per (block b, query q). 8 waves x 16 key-rows each.
// Base geometric features scalar, then 3x (16x16) MLP layers via WMMA in LDS.
__global__ __launch_bounds__(256) void plm_kernel(
    const float* __restrict__ ref_pos, const float* __restrict__ atom_mask,
    const int* __restrict__ uid, const float* __restrict__ cqck,
    const float* __restrict__ Woff, const float* __restrict__ Winv,
    const float* __restrict__ Wvld, const float* __restrict__ Wp1,
    const float* __restrict__ Wp2, const float* __restrict__ Wp3,
    float* __restrict__ plm_out) {
  __shared__ __align__(16) _Float16 ldsA[8][16 * 32];    // per-wave, K padded
  __shared__ __align__(16) float    ldsBase[8][16 * 16]; // residual / out tile
  __shared__ __align__(16) _Float16 ldsW[3][16 * 32];    // TRANSPOSED [col][k]

  const int tid = threadIdx.x, wave = tid >> 5, lane = tid & 31;
  const int b = blockIdx.x >> 5, q = blockIdx.x & 31;

  for (int i = tid; i < 3 * 512; i += 256) {
    int ws = i >> 9, rem = i & 511, c = rem >> 5, k = rem & 31;
    const float* Wsrc = (ws == 0) ? Wp1 : (ws == 1 ? Wp2 : Wp3);
    ldsW[ws][c * 32 + k] = (k < 16) ? (_Float16)Wsrc[k * 16 + c] : (_Float16)0.f;
  }

  const int qatom = b * NQ + q;
  const int kk = (wave << 4) + (lane & 15);      // key row within window
  const int ch0 = (lane < 16) ? 0 : 8;           // channel half this lane fills
  const int raw = b * NQ - 48 + kk;
  const int idx = min(max(raw, 0), N_ATOM - 1);
  const float vk = (raw >= 0 && raw < N_ATOM) ? 1.f : 0.f;
  const float pm = atom_mask[qatom] * atom_mask[idx] * vk;
  const float dx = (ref_pos[qatom * 3 + 0] - ref_pos[idx * 3 + 0]) * pm;
  const float dy = (ref_pos[qatom * 3 + 1] - ref_pos[idx * 3 + 1]) * pm;
  const float dz = (ref_pos[qatom * 3 + 2] - ref_pos[idx * 3 + 2]) * pm;
  const float vv = (uid[qatom] == uid[idx]) ? pm : 0.f;
  const float inv_sq = 1.f / (1.f + dx * dx + dy * dy + dz * dz);
  const float* cqr = cqck + (size_t)qatom * 32;       // relu(cl_q @ W_cq)
  const float* ckr = cqck + (size_t)idx * 32 + 16;    // relu(cl_k @ W_ck)
  const int rloc = lane & 15;

#pragma unroll
  for (int j = 0; j < 8; ++j) {
    int c = ch0 + j;
    float base = (dx * Woff[c] + dy * Woff[16 + c] + dz * Woff[32 + c]) * vv
               + inv_sq * Winv[c] * vv + Wvld[c] * vv * vv + cqr[c] + ckr[c];
    ldsBase[wave][rloc * 16 + c] = base;
    ldsA[wave][rloc * 32 + c] = (_Float16)fmaxf(base, 0.f);
    ldsA[wave][rloc * 32 + 16 + c] = (_Float16)0.f;  // K padding
  }
  __syncthreads();

  const int col = lane & 15, rb = (lane < 16) ? 0 : 8;
  _Float16* Aw = ldsA[wave];

  v8f acc1 = {0.f, 0.f, 0.f, 0.f, 0.f, 0.f, 0.f, 0.f};
  acc1 = wmma_f16(frag_a_h(Aw, 32, 0, 0, lane), frag_b_t(ldsW[0], 32, 0, 0, lane), acc1);
#pragma unroll
  for (int r = 0; r < 8; ++r)
    Aw[(rb + r) * 32 + col] = (_Float16)fmaxf(acc1[r], 0.f);

  v8f acc2 = {0.f, 0.f, 0.f, 0.f, 0.f, 0.f, 0.f, 0.f};
  acc2 = wmma_f16(frag_a_h(Aw, 32, 0, 0, lane), frag_b_t(ldsW[1], 32, 0, 0, lane), acc2);
#pragma unroll
  for (int r = 0; r < 8; ++r)
    Aw[(rb + r) * 32 + col] = (_Float16)fmaxf(acc2[r], 0.f);

  v8f acc3 = {0.f, 0.f, 0.f, 0.f, 0.f, 0.f, 0.f, 0.f};
  acc3 = wmma_f16(frag_a_h(Aw, 32, 0, 0, lane), frag_b_t(ldsW[2], 32, 0, 0, lane), acc3);

  // residual add, then restage tile and store as coalesced float4s
  float vals[8];
#pragma unroll
  for (int r = 0; r < 8; ++r)
    vals[r] = ldsBase[wave][(rb + r) * 16 + col] + acc3[r];
#pragma unroll
  for (int r = 0; r < 8; ++r)
    ldsBase[wave][(rb + r) * 16 + col] = vals[r];
  float* tile = plm_out + (((size_t)(b * NQ + q)) * NK + (wave << 4)) * 16;
#pragma unroll
  for (int qd = lane; qd < 64; qd += 32)
    *(float4*)(tile + qd * 4) = *(const float4*)(&ldsBase[wave][qd * 4]);
}

// ------------------------- local attention kernel --------------------------
// One workgroup (4 waves) per (block b, head h). 32 queries x 128 keys.
// Q/K staged via async global->LDS b128 when the toolchain exposes it.
__global__ __launch_bounds__(128) void attn_kernel(
    const _Float16* __restrict__ q16, const _Float16* __restrict__ k16,
    const _Float16* __restrict__ v16, const float* __restrict__ gw,
    const float* __restrict__ plm, const float* __restrict__ atom_mask,
    const float* __restrict__ Wpb, float* __restrict__ og) {
  __shared__ __align__(16) _Float16 ldsQ[NQ * CH];   // [q][c]
  __shared__ __align__(16) _Float16 ldsK[NK * CH];   // [key][c] (B^T layout)
  __shared__ __align__(16) _Float16 ldsVt[CH * NK];  // [c][key] (B^T for P.V)
  __shared__ __align__(16) float    ldsS[NQ * NK];
  __shared__ __align__(16) float    ldsO[4][256];
  __shared__ float wpb[16];

  const int b = blockIdx.x, h = blockIdx.y;
  const int tid = threadIdx.x, wave = tid >> 5, lane = tid & 31;
  if (tid < 16) wpb[tid] = Wpb[tid * 4 + h];

  const int j = tid;                       // one key per thread
  const int rawj = b * NQ - 48 + j;
  const int idxj = min(max(rawj, 0), N_ATOM - 1);
  const _Float16* kr = k16 + (size_t)idxj * 128 + h * CH;
  const _Float16* vr = v16 + (size_t)idxj * 128 + h * CH;
  const int qrow = tid >> 2, qpart = (tid & 3) << 3;
  const _Float16* qsrc = q16 + (size_t)(b * NQ + qrow) * 128 + h * CH + qpart;

#if HAVE_ASYNC_LDS
  // Q: 1x b128 per thread; K: 4x b128 per thread (ASYNCcnt-tracked)
  async_b128(qsrc, ldsQ + qrow * CH + qpart);
#pragma unroll
  for (int c = 0; c < CH; c += 8) async_b128(kr + c, ldsK + j * CH + c);
#else
  *(uint4*)(ldsQ + qrow * CH + qpart) = *(const uint4*)qsrc;
#pragma unroll
  for (int c = 0; c < CH; c += 8)
    *(uint4*)(ldsK + j * CH + c) = *(const uint4*)(kr + c);
#endif
  // V must land transposed -> manual scatter
#pragma unroll
  for (int c = 0; c < CH; c += 4) {
    h4 vvh = *(const h4*)(vr + c);
    ldsVt[(c + 0) * NK + j] = vvh.x;
    ldsVt[(c + 1) * NK + j] = vvh.y;
    ldsVt[(c + 2) * NK + j] = vvh.z;
    ldsVt[(c + 3) * NK + j] = vvh.w;
  }
#if HAVE_ASYNC_LDS
  __builtin_amdgcn_s_wait_asynccnt(0);
#endif
  __syncthreads();

  const float scale = 0.17677669529663687f;  // 1/sqrt(32)

  // scores S = Q K^T : 2x8 = 16 tiles over 4 waves, single K-step (CH=32)
  for (int t = wave; t < 16; t += 4) {
    int mt = t & 1, nt = t >> 1;
    int m0 = mt * 16, n0 = nt * 16;
    v16h af = frag_a_h(ldsQ, 32, m0, 0, lane);
    v16h bf = frag_b_t(ldsK, CH, 0, n0, lane);
    v8f acc = {0.f, 0.f, 0.f, 0.f, 0.f, 0.f, 0.f, 0.f};
    acc = wmma_f16(af, bf, acc);

    int col = n0 + (lane & 15);
    int rb = (lane < 16) ? 0 : 8;
    int raw = b * NQ - 48 + col;
    float vkm = (raw >= 0 && raw < N_ATOM)
                    ? atom_mask[min(max(raw, 0), N_ATOM - 1)] : 0.f;
#pragma unroll
    for (int r = 0; r < 8; ++r) {
      int qq = m0 + rb + r;
      const float4* p4 =
          (const float4*)(plm + (((size_t)b * NQ + qq) * NK + col) * 16);
      float bias = 0.f;
#pragma unroll
      for (int c4 = 0; c4 < 4; ++c4) {
        float4 pv = p4[c4];
        bias += pv.x * wpb[c4 * 4 + 0] + pv.y * wpb[c4 * 4 + 1] +
                pv.z * wpb[c4 * 4 + 2] + pv.w * wpb[c4 * 4 + 3];
      }
      float pmask = atom_mask[b * NQ + qq] * vkm;
      ldsS[qq * NK + col] = acc[r] * scale + bias + (1.f - pmask) * (-1e9f);
    }
  }
  __syncthreads();

  if (tid < NQ) {  // row softmax
    float mx = -1e30f;
    for (int k = 0; k < NK; ++k) mx = fmaxf(mx, ldsS[tid * NK + k]);
    float s = 0.f;
    for (int k = 0; k < NK; ++k) {
      float e = __expf(ldsS[tid * NK + k] - mx);
      ldsS[tid * NK + k] = e;
      s += e;
    }
    float inv = 1.f / s;
    for (int k = 0; k < NK; ++k) ldsS[tid * NK + k] *= inv;
  }
  __syncthreads();

  {  // O = P V : 2x2 tiles, one per wave, K = 128 in 4 WMMA steps
    int mt = wave & 1, nt = wave >> 1;
    int m0 = mt * 16, n0 = nt * 16;
    v8f acc = {0.f, 0.f, 0.f, 0.f, 0.f, 0.f, 0.f, 0.f};
    for (int k0 = 0; k0 < NK; k0 += 32) {
      v16h af = frag_a_f32(ldsS, NK, m0, k0, lane);
      v16h bf = frag_b_t(ldsVt, NK, k0, n0, lane);
      acc = wmma_f16(af, bf, acc);
    }
    // stage per-wave 16x16 tile, then gate + store as coalesced float4s
    int col = lane & 15, rb = (lane < 16) ? 0 : 8;
#pragma unroll
    for (int r = 0; r < 8; ++r) ldsO[wave][(rb + r) * 16 + col] = acc[r];
#pragma unroll
    for (int qd = lane; qd < 64; qd += 32) {
      int row = qd >> 2, part = (qd & 3) << 2;
      float4 o4 = *(const float4*)(&ldsO[wave][row * 16 + part]);
      size_t base = (size_t)(b * NQ + m0 + row) * 128 + h * CH + n0 + part;
      float4 g4 = *(const float4*)(gw + base);   // sigmoid applied in GEMM
      o4.x *= g4.x; o4.y *= g4.y; o4.z *= g4.z; o4.w *= g4.w;
      *(float4*)(og + base) = o4;
    }
  }
}

// ------------------------- misc elementwise --------------------------------
__global__ void mul_inplace(float* a, const float* __restrict__ b, int n) {
  int i = blockIdx.x * blockDim.x + threadIdx.x;
  if (i < n) a[i] *= b[i];
}

// Deterministic segment mean: atom_to_token is sorted -> binary-search range.
__global__ __launch_bounds__(128) void tok_agg_kernel(
    const float* __restrict__ feat, const int* __restrict__ a2t,
    float* __restrict__ tok) {
  const int t = blockIdx.x;
  int lo, hi;
  { int l = 0, r = N_ATOM; while (l < r) { int m = (l + r) >> 1; if (a2t[m] < t) l = m + 1; else r = m; } lo = l; }
  { int l = 0, r = N_ATOM; while (l < r) { int m = (l + r) >> 1; if (a2t[m] <= t) l = m + 1; else r = m; } hi = l; }
  float inv = 1.f / fmaxf((float)(hi - lo), 1.f);
  for (int c = threadIdx.x; c < 384; c += blockDim.x) {
    float s = 0.f;
    for (int r = lo; r < hi; ++r) s += feat[(size_t)r * 384 + c];
    tok[(size_t)t * 384 + c] = s * inv;
  }
}

// ------------------------- host orchestration ------------------------------
static inline int gemm_grid(int M, int N) { return (M / 128) * (N / 32); }

extern "C" void kernel_launch(void* const* d_in, const int* in_sizes, int n_in,
                              void* d_out, int out_size, void* d_ws, size_t ws_size,
                              hipStream_t stream) {
  (void)in_sizes; (void)n_in; (void)out_size; (void)ws_size;

  const float* ref_pos       = (const float*)d_in[0];
  const float* ref_charge    = (const float*)d_in[1];
  const float* ref_mask      = (const float*)d_in[2];
  const float* atom_mask     = (const float*)d_in[3];
  const float* ref_element   = (const float*)d_in[4];
  const float* ref_chars     = (const float*)d_in[5];
  const float* W_ref_pos     = (const float*)d_in[6];
  const float* W_ref_charge  = (const float*)d_in[7];
  const float* W_ref_mask    = (const float*)d_in[8];
  const float* W_ref_element = (const float*)d_in[9];
  const float* W_ref_chars   = (const float*)d_in[10];
  const float* W_ref_offset  = (const float*)d_in[11];
  const float* W_inv_sq      = (const float*)d_in[12];
  const float* W_valid       = (const float*)d_in[13];
  const float* W_cq          = (const float*)d_in[14];
  const float* W_ck          = (const float*)d_in[15];
  const float* W_p1          = (const float*)d_in[16];
  const float* W_p2          = (const float*)d_in[17];
  const float* W_p3          = (const float*)d_in[18];
  const float* ln_a          = (const float*)d_in[19];
  const float* W_q           = (const float*)d_in[20];
  const float* W_k           = (const float*)d_in[21];
  const float* W_v           = (const float*)d_in[22];
  const float* W_pb          = (const float*)d_in[23];
  const float* W_g           = (const float*)d_in[24];
  const float* W_o           = (const float*)d_in[25];
  const float* ln_t          = (const float*)d_in[26];
  const float* W_ta          = (const float*)d_in[27];
  const float* W_tb          = (const float*)d_in[28];
  const float* W_to          = (const float*)d_in[29];
  const float* W_tok         = (const float*)d_in[30];
  const int*   ref_space_uid = (const int*)d_in[31];
  const int*   atom_to_token = (const int*)d_in[32];

  // outputs: tok [2048,384], ql [8192,128], cl [8192,128], plm [256,32,128,16]
  float* tok_out = (float*)d_out;
  float* ql_out  = tok_out + (size_t)N_TOKEN * 384;
  float* cl_out  = ql_out + (size_t)N_ATOM * 128;
  float* plm_out = cl_out + (size_t)N_ATOM * 128;

  // workspace layout (floats), phase-based reuse of regionA
  float* w = (float*)d_ws;
  size_t off = 0;
  float* regionA = w;                 off += 4194304;        // acat / ta+tb / feat
  float* wcat    = w + off;           off += (size_t)KCAT * 128;
  float* wqk_ws  = w + off;           off += 128 * 32;
  float* cqck_ws = w + off;           off += (size_t)N_ATOM * 32;
  float* a_ws    = w + off;           off += (size_t)N_ATOM * 128;
  float* g_ws    = w + off;           off += (size_t)N_ATOM * 128;
  float* og_ws   = w + off;           off += (size_t)N_ATOM * 128;
  _Float16* q16_ws = (_Float16*)(w + off); off += (size_t)N_ATOM * 64;
  _Float16* k16_ws = (_Float16*)(w + off); off += (size_t)N_ATOM * 64;
  _Float16* v16_ws = (_Float16*)(w + off); off += (size_t)N_ATOM * 64;
  float* acat    = regionA;
  float* ta_ws   = regionA;
  float* tb_ws   = regionA + (size_t)N_ATOM * 256;
  float* feat_ws = regionA;

  // ---- embedding: cl (and initial ql) from one concat GEMM ----
  fill_acat<<<(N_ATOM * KCAT + 255) / 256, 256, 0, stream>>>(
      ref_pos, ref_charge, ref_mask, ref_element, ref_chars, acat);
  fill_wcat<<<(KCAT * 128 + 255) / 256, 256, 0, stream>>>(
      W_ref_pos, W_ref_charge, W_ref_mask, W_ref_element, W_ref_chars, wcat);
  fill_wqk<<<(128 * 32 + 255) / 256, 256, 0, stream>>>(W_cq, W_ck, wqk_ws);
  gemm_wmma<0><<<gemm_grid(N_ATOM, 128), 256, 0, stream>>>(
      acat, wcat, nullptr, cl_out, ql_out, nullptr, N_ATOM, 128, KCAT);

  // ---- pair conditioning (fused cq|ck, N=32) + fused pair tensor ----
  gemm_wmma<1><<<gemm_grid(N_ATOM, 32), 256, 0, stream>>>(
      cl_out, wqk_ws, nullptr, cqck_ws, nullptr, nullptr, N_ATOM, 32, 128);
  plm_kernel<<<NBLOCKS * NQ, 256, 0, stream>>>(
      ref_pos, atom_mask, ref_space_uid, cqck_ws,
      W_ref_offset, W_inv_sq, W_valid, W_p1, W_p2, W_p3, plm_out);

  // ---- 3 transformer blocks ----
  for (int blk = 0; blk < NBLK; ++blk) {
    const size_t wq = (size_t)blk * 128 * 128;
    const size_t wt = (size_t)blk * 128 * 256;
    ln_kernel<<<N_ATOM / 8, 256, 0, stream>>>(ql_out, ln_a + blk * 128, a_ws);
    gemm_wmma<0><<<gemm_grid(N_ATOM, 128), 256, 0, stream>>>(
        a_ws, W_q + wq, nullptr, nullptr, nullptr, q16_ws, N_ATOM, 128, 128);
    gemm_wmma<0><<<gemm_grid(N_ATOM, 128), 256, 0, stream>>>(
        a_ws, W_k + wq, nullptr, nullptr, nullptr, k16_ws, N_ATOM, 128, 128);
    gemm_wmma<0><<<gemm_grid(N_ATOM, 128), 256, 0, stream>>>(
        a_ws, W_v + wq, nullptr, nullptr, nullptr, v16_ws, N_ATOM, 128, 128);
    gemm_wmma<2><<<gemm_grid(N_ATOM, 128), 256, 0, stream>>>(
        a_ws, W_g + wq, nullptr, g_ws, nullptr, nullptr, N_ATOM, 128, 128);
    attn_kernel<<<dim3(NBLOCKS, NHEAD), 128, 0, stream>>>(
        q16_ws, k16_ws, v16_ws, g_ws, plm_out, atom_mask, W_pb + blk * 16 * 4,
        og_ws);
    gemm_wmma<0><<<gemm_grid(N_ATOM, 128), 256, 0, stream>>>(
        og_ws, W_o + wq, ql_out, ql_out, nullptr, nullptr, N_ATOM, 128, 128);
    ln_kernel<<<N_ATOM / 8, 256, 0, stream>>>(ql_out, ln_t + blk * 128, a_ws);
    gemm_wmma<3><<<gemm_grid(N_ATOM, 256), 256, 0, stream>>>(
        a_ws, W_ta + wt, nullptr, ta_ws, nullptr, nullptr, N_ATOM, 256, 128);
    gemm_wmma<0><<<gemm_grid(N_ATOM, 256), 256, 0, stream>>>(
        a_ws, W_tb + wt, nullptr, tb_ws, nullptr, nullptr, N_ATOM, 256, 128);
    mul_inplace<<<(N_ATOM * 256 + 255) / 256, 256, 0, stream>>>(
        ta_ws, tb_ws, N_ATOM * 256);
    gemm_wmma<0><<<gemm_grid(N_ATOM, 128), 256, 0, stream>>>(
        ta_ws, W_to + wt, ql_out, ql_out, nullptr, nullptr, N_ATOM, 128, 256);
  }

  // ---- token aggregation (deterministic segment mean) ----
  gemm_wmma<1><<<gemm_grid(N_ATOM, 384), 256, 0, stream>>>(
      ql_out, W_tok, nullptr, feat_ws, nullptr, nullptr, N_ATOM, 384, 128);
  tok_agg_kernel<<<N_TOKEN, 128, 0, stream>>>(feat_ws, atom_to_token, tok_out);
}
